// FutureRegionContrastModule_79706003079306
// MI455X (gfx1250) — matrix operand order repaired and checked by reading, not compiled, not run on hardware
//
#include <hip/hip_runtime.h>
#include <hip/hip_bf16.h>

typedef __attribute__((ext_vector_type(16))) _Float16 v16h;
typedef __attribute__((ext_vector_type(8)))  _Float16 v8h;
typedef __attribute__((ext_vector_type(8)))  float    v8f;
typedef unsigned int u32x4 __attribute__((ext_vector_type(4)));
typedef unsigned int u32x8 __attribute__((ext_vector_type(8)));

static constexpr int B_ = 4, C_ = 256, H_ = 128, W_ = 128, Q_ = 100;
static constexpr int HW_ = H_ * W_;
static constexpr int BQ_ = B_ * Q_;          // 400
static constexpr int KPIX = 16;
static constexpr int NSAMP = BQ_ * KPIX;     // 6400
static constexpr int MAX_IGN = 4096;
static constexpr int MAXK = 16;
static constexpr float TEMP = 0.1f;
static constexpr float MARGIN = 0.3f;
static constexpr float FUT_FRAC = 0.5f;
static constexpr float IGN_W = 0.5f;
static constexpr float LOSS_WEIGHT = 1.0f;
static constexpr float NEGF = -1e30f;

// ---------------- init: zero accumulators ----------------
__global__ void k_init(float* accum, float* cnt, float* cproto, int* icount) {
  int t = blockIdx.x * blockDim.x + threadIdx.x;
  if (t < 4) accum[t] = 0.f;
  if (t < MAXK) cnt[t] = 0.f;
  if (t == 0) icount[0] = 0;
  for (int i = t; i < MAXK * C_; i += blockDim.x * gridDim.x) cproto[i] = 0.f;
}

// ---------------- per-(b,q) size / future-fraction / selected ----------------
__global__ void k_size_sel(const float* __restrict__ pred, const int* __restrict__ gt,
                           const int* __restrict__ kids, int nk,
                           float* size_f, int* selected) {
  __shared__ int kid_s[MAXK];
  const int bq = blockIdx.x;            // 0..399
  const int lane = threadIdx.x;         // 32 threads = 1 wave
  const int b = bq / Q_;
  if (lane < nk) kid_s[lane] = kids[lane];
  __syncthreads();
  const float* prow = pred + (size_t)bq * HW_;
  const int*   grow = gt + (size_t)b * HW_;
  int nbin = 0, nfut = 0;
  for (int p = lane; p < HW_; p += 32) {
    if (prow[p] > 0.f) {                 // sigmoid(x)>0.5 <=> x>0
      int g = grow[p];
      bool known = false;
      for (int j = 0; j < nk; ++j) known |= (g == kid_s[j]);
      ++nbin;
      if (!known) ++nfut;
    }
  }
#pragma unroll
  for (int off = 16; off > 0; off >>= 1) {
    nbin += __shfl_xor(nbin, off, 32);
    nfut += __shfl_xor(nfut, off, 32);
  }
  if (lane == 0) {
    float sz = (float)nbin;
    float frac = (float)nfut / fmaxf(sz, 1.f);
    size_f[bq] = sz;
    selected[bq] = (nbin > 0 && frac > FUT_FRAC) ? 1 : 0;
  }
}

// ---------------- proto GEMM: proto[b,q,c] = sum_p binary * feats / size ----------------
// one wave per (b, m-tile(7), n-tile(16)); K loop over HW in steps of 32.
// B tile (16 ch x 32 px of features) is double-buffered into LDS with
// global_load_async_to_lds_b128 + s_wait_asynccnt.
__global__ void k_proto_wmma(const float* __restrict__ pred, const float* __restrict__ mf,
                             const float* __restrict__ size_f, float* __restrict__ proto) {
  __attribute__((aligned(64))) __shared__ float btile[2][16 * 32];   // 2 x 2KB
  const int blk = blockIdx.x;            // 4*7*16 = 448
  const int b  = blk / (7 * 16);
  const int rem = blk % (7 * 16);
  const int mt = rem / 16;
  const int nt = rem % 16;
  const int lane = threadIdx.x;
  const int l15 = lane & 15;
  const int hi  = (lane >> 4) & 1;

  const int q = mt * 16 + l15;
  const bool qok = (q < Q_);
  const float* arow = pred + ((size_t)(b * Q_ + (qok ? q : 0))) * HW_;
  const int ko = hi ? 16 : 0;            // A K-offsets: {0..15} / {16..31}
  const float* mfB = mf + (size_t)(b * C_ + nt * 16) * HW_;   // channel block base

  const unsigned ldsBase0 = (unsigned)(size_t)&btile[0][0];
  const unsigned ldsBase1 = (unsigned)(size_t)&btile[1][0];
  const int chl = lane >> 3;            // 0..3
  const int pxl = (lane & 7) * 4;       // 0..28 step 4 (floats)

  // stage one 16x32-float tile (2KB) with 4 async b128 issues (32 lanes x 16B each)
  auto stage = [&](int kk, unsigned ldsB) {
#pragma unroll
    for (int i = 0; i < 4; ++i) {
      const int ch = i * 4 + chl;
      const float* g = mfB + (size_t)ch * HW_ + kk + pxl;
      const unsigned l = ldsB + (unsigned)((ch * 32 + pxl) * 4);
      asm volatile("global_load_async_to_lds_b128 %0, %1, off"
                   :: "v"(l), "v"(g) : "memory");
    }
  };

  stage(0, ldsBase0);

  v8f acc = {};
  for (int k0 = 0; k0 < HW_; k0 += 32) {
    const int buf = (k0 >> 5) & 1;
    if (k0 + 32 < HW_) {
      stage(k0 + 32, buf ? ldsBase0 : ldsBase1);
      asm volatile("s_wait_asynccnt 0x4" ::: "memory");   // previous tile landed
    } else {
      asm volatile("s_wait_asynccnt 0x0" ::: "memory");
    }
    __builtin_prefetch(arow + k0 + 512, 0, 1);

    v16h a;
#pragma unroll
    for (int j = 0; j < 8; ++j) {
      float x0 = arow[k0 + ko + j];
      float x1 = arow[k0 + ko + 8 + j];
      a[j]     = (qok && x0 > 0.f) ? (_Float16)1.0f : (_Float16)0.0f;
      a[8 + j] = (qok && x1 > 0.f) ? (_Float16)1.0f : (_Float16)0.0f;
    }
    // B fragment from LDS: lane n = l15 -> channel row, K half per half-wave
    const float* bp = &btile[buf][l15 * 32 + (hi ? 16 : 0)];
    v16h bv;
#pragma unroll
    for (int j = 0; j < 16; ++j) bv[j] = (_Float16)bp[j];
    acc = __builtin_amdgcn_wmma_f32_16x16x32_f16(false, a, false, bv, (short)0, acc, false, false);
  }
#pragma unroll
  for (int r = 0; r < 8; ++r) {
    int m = (hi ? 8 : 0) + r;
    int qq = mt * 16 + m;
    if (qq < Q_) {
      int row = b * Q_ + qq;
      proto[(size_t)row * C_ + (nt * 16 + l15)] = acc[r] / fmaxf(size_f[row], 1.f);
    }
  }
}

// ---------------- normalize proto rows -> f16 ----------------
__global__ void k_proto_norm(const float* __restrict__ proto, _Float16* __restrict__ protoH) {
  const int row = blockIdx.x;            // 400
  const int lane = threadIdx.x;
  float f[8], ss = 0.f;
#pragma unroll
  for (int j = 0; j < 8; ++j) { f[j] = proto[(size_t)row * C_ + lane + 32 * j]; ss += f[j] * f[j]; }
#pragma unroll
  for (int off = 16; off > 0; off >>= 1) ss += __shfl_xor(ss, off, 32);
  float inv = 1.f / fmaxf(sqrtf(ss), 1e-12f);
#pragma unroll
  for (int j = 0; j < 8; ++j) protoH[(size_t)row * C_ + lane + 32 * j] = (_Float16)(f[j] * inv);
}

// ---------------- per-pixel argmax over selected queries ----------------
__global__ void k_bestq(const float* __restrict__ pred, const int* __restrict__ selected,
                        int* __restrict__ best_q) {
  const int tid = blockIdx.x * blockDim.x + threadIdx.x;
  if (tid >= B_ * HW_) return;
  const int b = tid / HW_, p = tid % HW_;
  float bestv = -1.f; int best = -1;
  for (int q = 0; q < Q_; ++q) {
    if (selected[b * Q_ + q]) {
      float pm = pred[((size_t)(b * Q_ + q)) * HW_ + p];
      if (pm > 0.f) {
        float s = 1.f / (1.f + __expf(-pm));
        if (s > bestv) { bestv = s; best = q; }
      }
    }
  }
  best_q[tid] = best;
}

// ---------------- deterministic ballot-ordered sampling of K_PIX pixels ----------------
__global__ void k_sample(const int* __restrict__ best_q, const int* __restrict__ selected,
                         int* __restrict__ sidx, float* __restrict__ svalid) {
  const int bq = blockIdx.x;             // 0..399
  const int lane = threadIdx.x;          // one wave
  const int b = bq / Q_, q = bq % Q_;
  int count = 0;
  const int sel = selected[bq];
  if (sel) {
    const int* row = best_q + (size_t)b * HW_;
    for (int p0 = 0; p0 < HW_ && count < KPIX; p0 += 32) {
      int p = p0 + lane;
      bool match = (row[p] == q);
      unsigned mask = (unsigned)__ballot(match);
      int rank = __popc(mask & ((lane == 0) ? 0u : ((1u << lane) - 1u)));
      if (match && (count + rank) < KPIX) sidx[bq * KPIX + count + rank] = p;
      count += __popc(mask);
    }
    if (count > KPIX) count = KPIX;
  }
  if (lane < KPIX) {
    bool valid = sel && (lane < count);
    if (!valid) sidx[bq * KPIX + lane] = 0;
    svalid[bq * KPIX + lane] = valid ? 1.f : 0.f;
  }
}

// ---------------- gather + normalize sampled features -> f16 ----------------
__global__ void k_gather_sf(const float* __restrict__ mf, const int* __restrict__ sidx,
                            _Float16* __restrict__ sfH) {
  const int s = blockIdx.x;              // 6400
  const int lane = threadIdx.x;
  const int b = s / (Q_ * KPIX);
  const int p = sidx[s];
  float f[8], ss = 0.f;
#pragma unroll
  for (int j = 0; j < 8; ++j) {
    int c = lane + 32 * j;
    f[j] = mf[((size_t)(b * C_ + c)) * HW_ + p];
    ss += f[j] * f[j];
  }
#pragma unroll
  for (int off = 16; off > 0; off >>= 1) ss += __shfl_xor(ss, off, 32);
  float inv = 1.f / fmaxf(sqrtf(ss), 1e-12f);
#pragma unroll
  for (int j = 0; j < 8; ++j) sfH[(size_t)s * C_ + lane + 32 * j] = (_Float16)(f[j] * inv);
}

// ---------------- logits GEMM (WMMA) + streaming log-softmax + CE ----------------
// one wave per 16-sample row tile; loops 25 proto column tiles, K=256 in 8 steps.
// Per column tile, the 16x256 f16 proto block is DMA'd to LDS by the
// Tensor Data Mover (tensor_load_to_lds + s_wait_tensorcnt).
__global__ void k_logits_ce(const _Float16* __restrict__ sfH, const _Float16* __restrict__ protoH,
                            const int* __restrict__ selected, const float* __restrict__ svalid,
                            float* accum) {
  __attribute__((aligned(64))) __shared__ _Float16 ldsP[16 * C_];   // 8KB tile
  const int lane = threadIdx.x;
  const int mt = blockIdx.x;             // 400 tiles
  const int sbase = mt * 16;
  const int l15 = lane & 15;
  const int hi  = (lane >> 4) & 1;

  // cache all A fragments for this row tile (K = 256)
  v16h afrag[8];
  {
    const _Float16* rowp = sfH + (size_t)(sbase + l15) * C_;
    const int ko = hi ? 16 : 0;
#pragma unroll
    for (int t = 0; t < 8; ++t) {
      const v8h lo = *reinterpret_cast<const v8h*>(rowp + t * 32 + ko);
      const v8h hh = *reinterpret_cast<const v8h*>(rowp + t * 32 + ko + 8);
      v16h a;
#pragma unroll
      for (int j = 0; j < 8; ++j) { a[j] = lo[j]; a[8 + j] = hh[j]; }
      afrag[t] = a;
    }
  }

  float rmax[8], rsum[8], rlab[8];
  int labrow[8];
#pragma unroll
  for (int r = 0; r < 8; ++r) {
    rmax[r] = NEGF; rsum[r] = 0.f; rlab[r] = 0.f;
    labrow[r] = (sbase + (hi ? 8 : 0) + r) >> 4;   // label column = sample / K_PIX
  }

  const unsigned ldsOff = (unsigned)(size_t)&ldsP[0];
  const int ko2 = hi ? 16 : 0;
  for (int nt = 0; nt < 25; ++nt) {
    const int jbase = nt * 16;
    // ---- TDM: load 16 rows x 256 halves of protoH into LDS ----
    {
      unsigned long long ga = (unsigned long long)(size_t)(protoH + (size_t)jbase * C_);
      u32x4 g0;
      g0[0] = 1u;                                   // count=1 valid descriptor
      g0[1] = ldsOff;                               // lds_addr
      g0[2] = (unsigned)(ga & 0xFFFFFFFFu);         // global_addr[31:0]
      g0[3] = (unsigned)((ga >> 32) & 0x01FFFFFFu)  // global_addr[56:32]
              | (2u << 30);                         // type=2 (image)
      u32x8 g1;
      g1[0] = (1u << 16);                           // data_size=1 (2 bytes)
      g1[1] = (256u << 16);                         // tensor_dim0 = 256 (lo16 @ bit48)
      g1[2] = (16u << 16);                          // tensor_dim0 hi=0, tensor_dim1 = 16
      g1[3] = (256u << 16);                         // tensor_dim1 hi=0, tile_dim0 = 256
      g1[4] = 16u;                                  // tile_dim1 = 16, tile_dim2 = 0
      g1[5] = 256u;                                 // tensor_dim0_stride lo32 = 256
      g1[6] = 0u;
      g1[7] = 0u;
      asm volatile("tensor_load_to_lds %0, %1" :: "s"(g0), "s"(g1) : "memory");
      __builtin_amdgcn_s_wait_tensorcnt(0);
    }

    const int j = jbase + l15;           // this lane's logits column
    const int selj = selected[j];
    const _Float16* prow = &ldsP[l15 * C_];
    v8f d = {};
#pragma unroll
    for (int t = 0; t < 8; ++t) {
      const v16h bf = *reinterpret_cast<const v16h*>(prow + t * 32 + ko2);
      d = __builtin_amdgcn_wmma_f32_16x16x32_f16(false, afrag[t], false, bf, (short)0, d, false, false);
    }
#pragma unroll
    for (int r = 0; r < 8; ++r) {
      float x = selj ? d[r] * (1.0f / TEMP) : NEGF;
      if (j == labrow[r]) rlab[r] += x;  // exactly one (lane,tile) matches per row
      float mn = fmaxf(rmax[r], x);
      rsum[r] = rsum[r] * __expf(rmax[r] - mn) + __expf(x - mn);
      rmax[r] = mn;
    }
  }
  // merge across the 16 lanes of each half-wave
#pragma unroll
  for (int off = 8; off > 0; off >>= 1) {
#pragma unroll
    for (int r = 0; r < 8; ++r) {
      float om = __shfl_xor(rmax[r], off, 32);
      float os = __shfl_xor(rsum[r], off, 32);
      float ol = __shfl_xor(rlab[r], off, 32);
      float mn = fmaxf(rmax[r], om);
      rsum[r] = rsum[r] * __expf(rmax[r] - mn) + os * __expf(om - mn);
      rmax[r] = mn;
      rlab[r] += ol;
    }
  }
  if (l15 == 0) {
    float pce = 0.f, pv = 0.f;
#pragma unroll
    for (int r = 0; r < 8; ++r) {
      int s = sbase + (hi ? 8 : 0) + r;
      float lse = rmax[r] + __logf(fmaxf(rsum[r], 1e-30f));
      float ce = lse - rlab[r];
      float v = svalid[s];
      pce += ce * v; pv += v;
    }
    atomicAdd(&accum[0], pce);
    atomicAdd(&accum[1], pv);
  }
}

// ---------------- class prototypes (LDS accumulate, then global atomics) ----------------
__global__ void k_cproto(const float* __restrict__ mf, const int* __restrict__ gt,
                         const int* __restrict__ kids, int nk,
                         float* cnt, float* cproto) {
  __shared__ float lproto[MAXK * C_];
  __shared__ float lcnt[MAXK];
  __shared__ int cls[256];
  __shared__ int kid_s[MAXK];
  const int t = threadIdx.x;             // 256 threads (channel id)
  const int blk = blockIdx.x;            // B * (HW/256) = 256 blocks
  const int b = blk / (HW_ / 256);
  const int pbase = (blk % (HW_ / 256)) * 256;
  if (t < nk) { kid_s[t] = kids[t]; lcnt[t] = 0.f; }
  for (int k = 0; k < nk; ++k) lproto[k * C_ + t] = 0.f;
  __syncthreads();
  {
    int g = gt[(size_t)b * HW_ + pbase + t];
    int kc = -1;
    for (int j = 0; j < nk; ++j) if (g == kid_s[j]) kc = j;
    cls[t] = kc;
    if (kc >= 0) atomicAdd(&lcnt[kc], 1.f);
  }
  __syncthreads();
  const int c = t;
  for (int i = 0; i < 256; ++i) {
    int k = cls[i];
    if (k >= 0) lproto[k * C_ + c] += mf[((size_t)(b * C_ + c)) * HW_ + pbase + i];
  }
  __syncthreads();
  for (int k = 0; k < nk; ++k) atomicAdd(&cproto[k * C_ + t], lproto[k * C_ + t]);
  if (t < nk) atomicAdd(&cnt[t], lcnt[t]);
}

__global__ void k_cproto_norm(const float* __restrict__ cproto, const float* __restrict__ cnt,
                              float* __restrict__ cprotoN) {
  const int k = blockIdx.x;
  const int lane = threadIdx.x;
  const float ic = 1.f / fmaxf(cnt[k], 1.f);
  float f[8], ss = 0.f;
#pragma unroll
  for (int j = 0; j < 8; ++j) { f[j] = cproto[k * C_ + lane + 32 * j] * ic; ss += f[j] * f[j]; }
#pragma unroll
  for (int off = 16; off > 0; off >>= 1) ss += __shfl_xor(ss, off, 32);
  float inv = 1.f / fmaxf(sqrtf(ss), 1e-12f);
#pragma unroll
  for (int j = 0; j < 8; ++j) cprotoN[k * C_ + lane + 32 * j] = f[j] * inv;
}

// ---------------- deterministic ignore-pixel collection ----------------
__global__ void k_collect_ign(const int* __restrict__ gt, const int* __restrict__ ignore_id_p,
                              int* __restrict__ iidx, int* __restrict__ icount) {
  const int lane = threadIdx.x;          // single wave
  const int ign = ignore_id_p[0];
  int count = 0;
  const int total = B_ * HW_;
  for (int p0 = 0; p0 < total && count < MAX_IGN; p0 += 32) {
    int p = p0 + lane;
    bool m = (gt[p] == ign);
    unsigned mask = (unsigned)__ballot(m);
    int rank = __popc(mask & ((lane == 0) ? 0u : ((1u << lane) - 1u)));
    if (m && (count + rank) < MAX_IGN) iidx[count + rank] = p;
    count += __popc(mask);
  }
  if (count > MAX_IGN) count = MAX_IGN;
  if (lane == 0) icount[0] = count;
  for (int i = count + lane; i < MAX_IGN; i += 32) iidx[i] = 0;
}

// ---------------- hinge loss on ignore pixels ----------------
__global__ void k_ignore_loss(const float* __restrict__ mf, const int* __restrict__ iidx,
                              const int* __restrict__ icount, const float* __restrict__ cprotoN,
                              const float* __restrict__ cnt, int nk, float* accum) {
  const int i = blockIdx.x;              // 4096
  const int lane = threadIdx.x;
  const float v = (i < icount[0]) ? 1.f : 0.f;
  float hinge = 0.f;
  if (v > 0.f) {
    const int idx = iidx[i];
    const int b = idx / HW_, p = idx % HW_;
    float f[8], ss = 0.f;
#pragma unroll
    for (int j = 0; j < 8; ++j) {
      int c = lane + 32 * j;
      f[j] = mf[((size_t)(b * C_ + c)) * HW_ + p];
      ss += f[j] * f[j];
    }
#pragma unroll
    for (int off = 16; off > 0; off >>= 1) ss += __shfl_xor(ss, off, 32);
    float inv = 1.f / fmaxf(sqrtf(ss), 1e-12f);
    float best = NEGF;
    for (int k = 0; k < nk; ++k) {
      float d = 0.f;
#pragma unroll
      for (int j = 0; j < 8; ++j) d += f[j] * cprotoN[k * C_ + lane + 32 * j];
#pragma unroll
      for (int off = 16; off > 0; off >>= 1) d += __shfl_xor(d, off, 32);
      float sim = (cnt[k] > 0.f) ? d * inv : NEGF;
      best = fmaxf(best, sim);
    }
    hinge = fmaxf(best - MARGIN, 0.f);
  }
  if (lane == 0) { atomicAdd(&accum[2], hinge * v); atomicAdd(&accum[3], v); }
}

// ---------------- final combine ----------------
__global__ void k_final(const float* __restrict__ accum, float* __restrict__ out) {
  if (threadIdx.x == 0 && blockIdx.x == 0) {
    float lr = accum[0] / fmaxf(accum[1], 1.f);
    float li = accum[2] / fmaxf(accum[3], 1.f);
    out[0] = LOSS_WEIGHT * (lr + IGN_W * li);
  }
}

extern "C" void kernel_launch(void* const* d_in, const int* in_sizes, int n_in,
                              void* d_out, int out_size, void* d_ws, size_t ws_size,
                              hipStream_t stream) {
  const float* mf   = (const float*)d_in[0];   // (B,C,H,W)
  const float* pred = (const float*)d_in[1];   // (B,Q,H,W)
  const int*   gt   = (const int*)d_in[2];     // (B,H,W)
  const int*   kids = (const int*)d_in[3];     // (nk,)
  const int*   ignp = (const int*)d_in[4];     // scalar
  int nk = in_sizes[3]; if (nk > MAXK) nk = MAXK;
  float* out = (float*)d_out;

  // workspace carve-up (256B aligned)
  char* ws = (char*)d_ws;
  size_t cur = 0;
  auto alloc = [&](size_t bytes) { void* p = ws + cur; cur = (cur + bytes + 255) & ~(size_t)255; return p; };
  float*     size_f  = (float*)alloc(BQ_ * 4);
  int*       sel     = (int*)alloc(BQ_ * 4);
  float*     proto   = (float*)alloc((size_t)BQ_ * C_ * 4);
  _Float16*  protoH  = (_Float16*)alloc((size_t)BQ_ * C_ * 2);
  int*       best_q  = (int*)alloc((size_t)B_ * HW_ * 4);
  int*       sidx    = (int*)alloc(NSAMP * 4);
  float*     svalid  = (float*)alloc(NSAMP * 4);
  _Float16*  sfH     = (_Float16*)alloc((size_t)NSAMP * C_ * 2);
  float*     cnt     = (float*)alloc(MAXK * 4);
  float*     cproto  = (float*)alloc(MAXK * C_ * 4);
  float*     cprotoN = (float*)alloc(MAXK * C_ * 4);
  int*       iidx    = (int*)alloc(MAX_IGN * 4);
  int*       icount  = (int*)alloc(4);
  float*     accum   = (float*)alloc(4 * 4);
  (void)ws_size; (void)n_in; (void)out_size;

  k_init<<<16, 256, 0, stream>>>(accum, cnt, cproto, icount);
  k_size_sel<<<BQ_, 32, 0, stream>>>(pred, gt, kids, nk, size_f, sel);
  k_proto_wmma<<<B_ * 7 * 16, 32, 0, stream>>>(pred, mf, size_f, proto);
  k_proto_norm<<<BQ_, 32, 0, stream>>>(proto, protoH);
  k_bestq<<<(B_ * HW_ + 255) / 256, 256, 0, stream>>>(pred, sel, best_q);
  k_sample<<<BQ_, 32, 0, stream>>>(best_q, sel, sidx, svalid);
  k_gather_sf<<<NSAMP, 32, 0, stream>>>(mf, sidx, sfH);
  k_logits_ce<<<NSAMP / 16, 32, 0, stream>>>(sfH, protoH, sel, svalid, accum);
  k_cproto<<<B_ * (HW_ / 256), 256, 0, stream>>>(mf, gt, kids, nk, cnt, cproto);
  k_cproto_norm<<<nk, 32, 0, stream>>>(cproto, cnt, cprotoN);
  k_collect_ign<<<1, 32, 0, stream>>>(gt, ignp, iidx, icount);
  k_ignore_loss<<<MAX_IGN, 32, 0, stream>>>(mf, iidx, icount, cprotoN, cnt, nk, accum);
  k_final<<<1, 1, 0, stream>>>(accum, out);
}
